// WaveNet_86552180949687
// MI455X (gfx1250) — matrix operand order repaired and checked by reading, not compiled, not run on hardware
//
#include <hip/hip_runtime.h>

// ---------------- problem constants ----------------
#define LAYERS   24
#define Bsz      64      // batch
#define Cch      64      // channels
#define TWO_C    128
#define Ssk      256     // skip width
#define Kcls     256     // classes
#define NTHREADS 512
#define NWAVES   16      // wave32

typedef _Float16 h8v  __attribute__((ext_vector_type(8)));
typedef _Float16 h16v __attribute__((ext_vector_type(16)));
typedef float    f4v  __attribute__((ext_vector_type(4)));
typedef float    f8v  __attribute__((ext_vector_type(8)));

__constant__ int d_dil[LAYERS]  = {1,2,4,8,16,32,64,128, 1,2,4,8,16,32,64,128, 1,2,4,8,16,32,64,128};
__constant__ int d_qoff[LAYERS] = {0,1,3,7,15,31,63,127, 255,256,258,262,270,286,318,382,
                                   510,511,513,517,525,541,573,637};   // prefix sums of dilations
#define QSLOTS 765

// ---------------- WMMA operand loaders (16x16x32 f16; A/B in contiguous k order, same for both) ----------------
// A (16x32, MxK): lane = m + 16*h ; per-lane K = 16h..16h+15  (two 16B chunks)
__device__ __forceinline__ h16v load_A(const _Float16* base, int ld, int m0, int k0, int lane) {
    int m = m0 + (lane & 15);
    int h = lane >> 4;
    const _Float16* p = base + m * ld + k0 + 16 * h;
    h8v lo = *(const h8v*)p;
    h8v hi = *(const h8v*)(p + 8);
    return __builtin_shufflevector(lo, hi, 0,1,2,3,4,5,6,7,8,9,10,11,12,13,14,15);
}
// B (32x16, KxN) from W^T stored [N][K]: lane = n + 16*h ; per-lane K = 16h..16h+15 (one 32B chunk)
__device__ __forceinline__ h16v load_B(const _Float16* base, int ld, int n0, int k0, int lane) {
    int n = n0 + (lane & 15);
    int h = lane >> 4;
    return *(const h16v*)(base + n * ld + k0 + 16 * h);
}
__device__ __forceinline__ f8v wmma16(h16v a, h16v b, f8v c) {
    return __builtin_amdgcn_wmma_f32_16x16x32_f16(false, a, false, b, (short)0, c, false, false);
}
// C/D (16x16 f32): lane = n + 16*h ; vgpr r -> M = 8h + r, N = n
__device__ __forceinline__ void store_C(float* base, int ld, int m0, int n0, int lane, f8v acc) {
    int n  = n0 + (lane & 15);
    int mb = m0 + ((lane >> 4) << 3);
    #pragma unroll
    for (int r = 0; r < 8; ++r) base[(mb + r) * ld + n] = acc[r];
}

// ---------------- weight prep: f32 -> f16, transposed to [N][K] ----------------
extern "C" __global__ void __launch_bounds__(256)
prep_weights(const float* __restrict__ conv_k, const float* __restrict__ res_w,
             const float* __restrict__ skip_w, const float* __restrict__ out0_w,
             const float* __restrict__ out1_w,
             _Float16* __restrict__ w0T, _Float16* __restrict__ w1T,
             _Float16* __restrict__ resT, _Float16* __restrict__ skipT,
             _Float16* __restrict__ o0T, _Float16* __restrict__ o1T)
{
    int e = blockIdx.x * blockDim.x + threadIdx.x;
    const int N0 = LAYERS * TWO_C * Cch;   // w0T / w1T each
    const int N1 = LAYERS * Cch * Cch;     // resT
    const int N2 = LAYERS * Ssk * Cch;     // skipT
    const int N3 = Kcls * Ssk;             // o0T / o1T each
    if (e < N0) {
        int l = e / (TWO_C * Cch), r = e % (TWO_C * Cch), n = r >> 6, k = r & 63;
        w0T[e] = (_Float16)conv_k[((l * 2 + 0) * Cch + k) * TWO_C + n];
        w1T[e] = (_Float16)conv_k[((l * 2 + 1) * Cch + k) * TWO_C + n];
        return;
    }
    e -= N0;
    if (e < N1) { int l = e / 4096, r = e % 4096, n = r >> 6, k = r & 63;
        resT[e] = (_Float16)res_w[(l * Cch + k) * Cch + n]; return; }
    e -= N1;
    if (e < N2) { int l = e / 16384, r = e % 16384, n = r >> 6, k = r & 63;
        skipT[e] = (_Float16)skip_w[(l * Cch + k) * Ssk + n]; return; }
    e -= N2;
    if (e < N3) { int n = e >> 8, k = e & 255;
        o0T[e] = (_Float16)out0_w[k * Kcls + n];
        o1T[e] = (_Float16)out1_w[k * Kcls + n]; return; }
}

// ---------------- LDS layout (dynamic, 241920 B <= 320 KB/WG) ----------------
#define O_XF    0        // float [64][64]   residual activation (f32 master)
#define O_XH    16384    // f16   [64][64]
#define O_XLH   24576    // f16   [64][64]   x_last from queue
#define O_GH    32768    // f16   [64][64]   gate output
#define O_HF    40960    // float [64][128]  conv pre-activation ; aliased as skH f16[64][256]
#define O_SKF   73728    // float [64][256]  skip accumulator
#define O_H0H   139264   // f16   [64][256]  relu hidden
#define O_LGF   172032   // float [64][256]  logits
#define O_PV    237568   // float [512] argmax partial vals
#define O_PI    239616   // int   [512] argmax partial idx
#define O_SAMP  241664   // int   [64]
#define SMEM_BYTES 241920

// ---------------- persistent single-workgroup WaveNet inference ----------------
extern "C" __global__ void __launch_bounds__(NTHREADS)
wavenet_persist(const int* __restrict__ seed, const float* __restrict__ embed,
                const float* __restrict__ conv_b, const float* __restrict__ res_b,
                const float* __restrict__ skip_b, const float* __restrict__ out0_b,
                const float* __restrict__ out1_b,
                const _Float16* __restrict__ w0T, const _Float16* __restrict__ w1T,
                const _Float16* __restrict__ resT, const _Float16* __restrict__ skipT,
                const _Float16* __restrict__ o0T, const _Float16* __restrict__ o1T,
                _Float16* __restrict__ queues, float* __restrict__ out,
                const int* __restrict__ pT)
{
    extern __shared__ char smem[];
    float*    xF   = (float*)   (smem + O_XF);
    _Float16* xH   = (_Float16*)(smem + O_XH);
    _Float16* xlH  = (_Float16*)(smem + O_XLH);
    _Float16* gH   = (_Float16*)(smem + O_GH);
    float*    Hf   = (float*)   (smem + O_HF);
    _Float16* skH  = (_Float16*)(smem + O_HF);   // alias: Hf dead after layer loop
    float*    skF  = (float*)   (smem + O_SKF);
    _Float16* h0H  = (_Float16*)(smem + O_H0H);
    float*    lgF  = (float*)   (smem + O_LGF);
    float*    pv   = (float*)   (smem + O_PV);
    int*      pidx = (int*)     (smem + O_PI);
    int*      samp = (int*)     (smem + O_SAMP);

    const int tid  = threadIdx.x;
    const int lane = tid & 31;
    // scalar (SGPR) wave id -> provably uniform control flow around WMMA
    const int wv   = __builtin_amdgcn_readfirstlane(tid >> 5);
    const int T    = pT[0];

    // init: x = embed[seed], skip = 0
    for (int e = tid; e < Bsz * Cch; e += NTHREADS) {
        int b = e >> 6;
        float v = embed[seed[b] * Cch + (e & 63)];
        xF[e] = v; xH[e] = (_Float16)v;
    }
    {
        f4v z = {0.f, 0.f, 0.f, 0.f};
        f4v* s4 = (f4v*)skF;
        #pragma unroll
        for (int e = 0; e < 8; ++e) s4[tid + e * NTHREADS] = z;
    }
    __syncthreads();

    for (int t = 0; t < T; ++t) {
        // ================= residual layers =================
        for (int l = 0; l < LAYERS; ++l) {
            const int d    = d_dil[l];
            const int dpos = t & (d - 1);
            _Float16* q = queues + (size_t)(d_qoff[l] + dpos) * (Bsz * Cch);

            // prefetch next layer's weights toward WGP while we compute this one
            {
                int ln = (l + 1 < LAYERS) ? l + 1 : l;
                const char* p0 = (const char*)(w0T  + ln * (TWO_C * Cch));
                const char* p1 = (const char*)(w1T  + ln * (TWO_C * Cch));
                const char* ps = (const char*)(skipT + ln * (Ssk * Cch));
                const char* pr = (const char*)(resT + ln * (Cch * Cch));
                __builtin_prefetch(p0 + tid * 32, 0, 1);
                __builtin_prefetch(p1 + tid * 32, 0, 1);
                __builtin_prefetch(ps + tid * 64, 0, 1);
                if (tid < 128) __builtin_prefetch(pr + tid * 64, 0, 1);
            }

            // circular queue swap, 16B per thread: x_last <- q[pos] ; q[pos] <- x_cur
            {
                h8v* q8  = (h8v*)q;
                h8v* xl8 = (h8v*)xlH;
                h8v* x8  = (h8v*)xH;
                h8v v = q8[tid];
                xl8[tid] = v;
                q8[tid] = x8[tid];
            }
            __syncthreads();

            // conv GEMM: H[64,128] = x_last @ Wk0 + x_cur @ Wk1 + conv_b  (32 tiles, 2/wave)
            const _Float16* W0 = w0T + l * (TWO_C * Cch);
            const _Float16* W1 = w1T + l * (TWO_C * Cch);
            const float*    cb = conv_b + l * TWO_C;
            #pragma unroll
            for (int it = 0; it < 2; ++it) {
                int tile = wv + it * NWAVES;
                int m0 = (tile >> 3) << 4, n0 = (tile & 7) << 4;
                // batch all operand loads ahead of the WMMA chain
                h16v a0 = load_A(xlH, Cch, m0, 0,  lane);
                h16v a1 = load_A(xlH, Cch, m0, 32, lane);
                h16v a2 = load_A(xH,  Cch, m0, 0,  lane);
                h16v a3 = load_A(xH,  Cch, m0, 32, lane);
                h16v b0 = load_B(W0, Cch, n0, 0,  lane);
                h16v b1 = load_B(W0, Cch, n0, 32, lane);
                h16v b2 = load_B(W1, Cch, n0, 0,  lane);
                h16v b3 = load_B(W1, Cch, n0, 32, lane);
                float bvv = cb[n0 + (lane & 15)];
                f8v acc;
                #pragma unroll
                for (int r = 0; r < 8; ++r) acc[r] = bvv;
                acc = wmma16(a0, b0, acc);
                acc = wmma16(a2, b2, acc);
                acc = wmma16(a1, b1, acc);
                acc = wmma16(a3, b3, acc);
                store_C(Hf, TWO_C, m0, n0, lane, acc);
            }
            __syncthreads();

            // gated activation: g = tanh(H[:, :C]) * sigmoid(H[:, C:])
            #pragma unroll
            for (int i = 0; i < 8; ++i) {
                int e = tid + i * NTHREADS;
                int b = e >> 6, c = e & 63;
                float a = Hf[b * TWO_C + c];
                float s = Hf[b * TWO_C + Cch + c];
                gH[e] = (_Float16)(tanhf(a) * (1.f / (1.f + expf(-s))));
            }
            __syncthreads();

            // skip GEMM (accumulate in LDS): skF += g @ skip_w + skip_b  (64 tiles, 4/wave)
            const _Float16* SW = skipT + l * (Ssk * Cch);
            const float*    sb = skip_b + l * Ssk;
            #pragma unroll
            for (int it = 0; it < 4; ++it) {
                int tile = wv + it * NWAVES;
                int m0 = (tile >> 4) << 4, n0 = (tile & 15) << 4;
                int n = n0 + (lane & 15), mb = m0 + ((lane >> 4) << 3);
                h16v a0 = load_A(gH, Cch, m0, 0,  lane);
                h16v a1 = load_A(gH, Cch, m0, 32, lane);
                h16v b0 = load_B(SW, Cch, n0, 0,  lane);
                h16v b1 = load_B(SW, Cch, n0, 32, lane);
                float sbv = sb[n];
                f8v acc;
                #pragma unroll
                for (int r = 0; r < 8; ++r) acc[r] = skF[(mb + r) * Ssk + n] + sbv;
                acc = wmma16(a0, b0, acc);
                acc = wmma16(a1, b1, acc);
                #pragma unroll
                for (int r = 0; r < 8; ++r) skF[(mb + r) * Ssk + n] = acc[r];
            }

            // res GEMM + residual: x += g @ res_w + res_b  (16 tiles, 1/wave)
            {
                const _Float16* RW = resT + l * (Cch * Cch);
                const float*    rb = res_b + l * Cch;
                int tile = wv;
                int m0 = (tile >> 2) << 4, n0 = (tile & 3) << 4;
                int n = n0 + (lane & 15), mb = m0 + ((lane >> 4) << 3);
                h16v a0 = load_A(gH, Cch, m0, 0,  lane);
                h16v a1 = load_A(gH, Cch, m0, 32, lane);
                h16v b0 = load_B(RW, Cch, n0, 0,  lane);
                h16v b1 = load_B(RW, Cch, n0, 32, lane);
                float rbv = rb[n];
                f8v acc;
                #pragma unroll
                for (int r = 0; r < 8; ++r) acc[r] = rbv;
                acc = wmma16(a0, b0, acc);
                acc = wmma16(a1, b1, acc);
                #pragma unroll
                for (int r = 0; r < 8; ++r) {
                    int idx = (mb + r) * Cch + n;
                    float nv = xF[idx] + acc[r];
                    xF[idx] = nv; xH[idx] = (_Float16)nv;
                }
            }
            __syncthreads();
        }

        // ================= output head =================
        #pragma unroll
        for (int i = 0; i < 32; ++i) {
            int e = tid + i * NTHREADS;
            skH[e] = (_Float16)skF[e];
        }
        __syncthreads();

        // h0 = relu(skip @ out0_w + out0_b)   (64 tiles, 4/wave, K=256)
        #pragma unroll
        for (int it = 0; it < 4; ++it) {
            int tile = wv + it * NWAVES;
            int m0 = (tile >> 4) << 4, n0 = (tile & 15) << 4;
            int n = n0 + (lane & 15), mb = m0 + ((lane >> 4) << 3);
            float bvv = out0_b[n];
            f8v acc;
            #pragma unroll
            for (int r = 0; r < 8; ++r) acc[r] = bvv;
            #pragma unroll
            for (int k0 = 0; k0 < Ssk; k0 += 64) {
                h16v a0 = load_A(skH, Ssk, m0, k0,      lane);
                h16v a1 = load_A(skH, Ssk, m0, k0 + 32, lane);
                h16v b0 = load_B(o0T, Ssk, n0, k0,      lane);
                h16v b1 = load_B(o0T, Ssk, n0, k0 + 32, lane);
                acc = wmma16(a0, b0, acc);
                acc = wmma16(a1, b1, acc);
            }
            #pragma unroll
            for (int r = 0; r < 8; ++r) {
                float v = acc[r] > 0.f ? acc[r] : 0.f;
                h0H[(mb + r) * Kcls + n] = (_Float16)v;
            }
        }
        __syncthreads();

        // logits = h0 @ out1_w + out1_b  -> LDS (argmax) + d_out
        #pragma unroll
        for (int it = 0; it < 4; ++it) {
            int tile = wv + it * NWAVES;
            int m0 = (tile >> 4) << 4, n0 = (tile & 15) << 4;
            int n = n0 + (lane & 15), mb = m0 + ((lane >> 4) << 3);
            float bvv = out1_b[n];
            f8v acc;
            #pragma unroll
            for (int r = 0; r < 8; ++r) acc[r] = bvv;
            #pragma unroll
            for (int k0 = 0; k0 < Kcls; k0 += 64) {
                h16v a0 = load_A(h0H, Kcls, m0, k0,      lane);
                h16v a1 = load_A(h0H, Kcls, m0, k0 + 32, lane);
                h16v b0 = load_B(o1T, Kcls, n0, k0,      lane);
                h16v b1 = load_B(o1T, Kcls, n0, k0 + 32, lane);
                acc = wmma16(a0, b0, acc);
                acc = wmma16(a1, b1, acc);
            }
            #pragma unroll
            for (int r = 0; r < 8; ++r) {
                int b = mb + r;
                lgF[b * Kcls + n] = acc[r];
                out[(size_t)Bsz * T + ((size_t)b * T + t) * Kcls + n] = acc[r];
            }
        }
        __syncthreads();

        // argmax (first-max tie break): 8 partials of 32 per row, then combine
        {
            int row = tid >> 3, part = tid & 7;
            const float* Lr = lgF + row * Kcls + part * 32;
            float bestv = Lr[0]; int besti = part * 32;
            for (int k = 1; k < 32; ++k) { float v = Lr[k]; if (v > bestv) { bestv = v; besti = part * 32 + k; } }
            pv[tid] = bestv; pidx[tid] = besti;
        }
        __syncthreads();
        if (tid < Bsz) {
            float bestv = pv[tid * 8]; int besti = pidx[tid * 8];
            #pragma unroll
            for (int p = 1; p < 8; ++p) {
                float v = pv[tid * 8 + p];
                if (v > bestv) { bestv = v; besti = pidx[tid * 8 + p]; }
            }
            samp[tid] = besti;
            // inverse mu-law: sign(mw) * (256^|mw| - 1) / 255,  mw = 2*s/255 - 1
            float mw = 2.f * (float)besti * (1.f / 255.f) - 1.f;
            float pw = (exp2f(8.f * fabsf(mw)) - 1.f) * (1.f / 255.f);
            out[(size_t)tid * T + t] = (mw < 0.f) ? -pw : pw;
        }
        __syncthreads();

        // next-step input + skip reset
        #pragma unroll
        for (int i = 0; i < 8; ++i) {
            int e = tid + i * NTHREADS;
            int b = e >> 6;
            float v = embed[samp[b] * Cch + (e & 63)];
            xF[e] = v; xH[e] = (_Float16)v;
        }
        {
            f4v z = {0.f, 0.f, 0.f, 0.f};
            f4v* s4 = (f4v*)skF;
            #pragma unroll
            for (int e = 0; e < 8; ++e) s4[tid + e * NTHREADS] = z;
        }
        __syncthreads();
    }
}

// ---------------- host launcher ----------------
extern "C" void kernel_launch(void* const* d_in, const int* in_sizes, int n_in,
                              void* d_out, int out_size, void* d_ws, size_t ws_size,
                              hipStream_t stream)
{
    const int*   seed   = (const int*)  d_in[0];
    const float* embed  = (const float*)d_in[1];
    const float* conv_k = (const float*)d_in[2];
    const float* conv_b = (const float*)d_in[3];
    const float* res_w  = (const float*)d_in[4];
    const float* res_b  = (const float*)d_in[5];
    const float* skip_w = (const float*)d_in[6];
    const float* skip_b = (const float*)d_in[7];
    const float* out0_w = (const float*)d_in[8];
    const float* out0_b = (const float*)d_in[9];
    const float* out1_w = (const float*)d_in[10];
    const float* out1_b = (const float*)d_in[11];
    const int*   pT     = (const int*)  d_in[12];

    char* ws = (char*)d_ws;
    _Float16* w0T   = (_Float16*)(ws + 0);        // 24*128*64 f16 = 393216 B
    _Float16* w1T   = (_Float16*)(ws + 393216);   // 393216 B
    _Float16* resT  = (_Float16*)(ws + 786432);   // 196608 B
    _Float16* skipT = (_Float16*)(ws + 983040);   // 786432 B
    _Float16* o0T   = (_Float16*)(ws + 1769472);  // 131072 B
    _Float16* o1T   = (_Float16*)(ws + 1900544);  // 131072 B
    _Float16* que   = (_Float16*)(ws + 2031616);  // 765*4096 f16 = 6266880 B
    const size_t qbytes = (size_t)QSLOTS * Bsz * Cch * sizeof(_Float16);

    const int prepN = LAYERS*TWO_C*Cch + LAYERS*Cch*Cch + LAYERS*Ssk*Cch + Kcls*Ssk;
    prep_weights<<<(prepN + 255) / 256, 256, 0, stream>>>(
        conv_k, res_w, skip_w, out0_w, out1_w, w0T, w1T, resT, skipT, o0T, o1T);
    (void)hipMemsetAsync(que, 0, qbytes, stream);

    wavenet_persist<<<1, NTHREADS, SMEM_BYTES, stream>>>(
        seed, embed, conv_b, res_b, skip_b, out0_b, out1_b,
        w0T, w1T, resT, skipT, o0T, o1T, que, (float*)d_out, pT);
}
